// LocalAttention_5368709120204
// MI455X (gfx1250) — matrix operand order repaired and checked by reading, not compiled, not run on hardware
//
#include <hip/hip_runtime.h>

// ---------------------------------------------------------------------------
// Fused Swin-window attention for MI455X (gfx1250, wave32, WMMA bf16).
// One workgroup (256 threads = 8 waves) per 7x7 window; one wave per head
// for the attention core. All intermediates live in LDS (~225 KB).
// ---------------------------------------------------------------------------

#define PW     7
#define PP     49            // tokens per window
#define HEADS  8
#define DHEAD  64
#define DIM    256
#define INNER  512           // HEADS * DHEAD
#define NQKV   1536          // INNER * 3
#define MPAD   64            // 49 padded up to 4 WMMA row-tiles
#define ATTN_SCALE 0.125f    // 64^-0.5

typedef __attribute__((ext_vector_type(16))) __bf16 v16bf;
typedef __attribute__((ext_vector_type(8)))  float  v8f;
typedef __attribute__((ext_vector_type(4)))  unsigned int v4u;

union FragU { v16bf v; v4u u[2]; };

__device__ __forceinline__ unsigned short f2bf(float x) {
  union { float f; unsigned u; } c; c.f = x;
  unsigned r = c.u + 0x7FFFu + ((c.u >> 16) & 1u);   // round-to-nearest-even
  return (unsigned short)(r >> 16);
}

__device__ __forceinline__ v8f vzero8() {
  v8f z = {0.f, 0.f, 0.f, 0.f, 0.f, 0.f, 0.f, 0.f};
  return z;
}

// A-fragment (16x32 bf16) from a row-major [rows][ld] matrix at (row0, k0).
// ISA layout: lanes 0-15 hold row M=lane, K in {0..7, 16..23} (+k0);
// lanes 16-31 hold the same rows, K in {8..15, 24..31}.
__device__ __forceinline__ v16bf load_a(const unsigned short* base, int ld,
                                        int row0, int k0, int lane) {
  int l15 = lane & 15, hi = lane >> 4;
  const unsigned short* p = base + (row0 + l15) * ld + k0 + hi * 8;
  FragU f;
  f.u[0] = *(const v4u*)(p);        // K = kb .. kb+7
  f.u[1] = *(const v4u*)(p + 16);   // K = kb+16 .. kb+23
  return f.v;
}

// B-fragment (32x16 bf16) from *transposed* storage BT[n][k] (row-major, ld=K
// stride). Layout matches load_a: lanes 0-15 hold column N=lane, K in {0..7, 16..23};
// lanes 16-31 hold K in {8..15, 24..31}.
__device__ __forceinline__ v16bf load_bt(const unsigned short* baseT, int ld,
                                         int n0, int k0, int lane) {
  int l15 = lane & 15, hi = lane >> 4;
  const unsigned short* p = baseT + (n0 + l15) * ld + k0 + hi * 8;
  FragU f;
  f.u[0] = *(const v4u*)(p);
  f.u[1] = *(const v4u*)(p + 16);
  return f.v;
}

__device__ __forceinline__ v8f wmma_bf16(v16bf a, v16bf b, v8f c) {
  return __builtin_amdgcn_wmma_f32_16x16x32_bf16(
      /*neg_a=*/false, a, /*neg_b=*/false, b,
      /*c_mod=*/(short)0, c, /*reuse_a=*/false, /*reuse_b=*/false);
}

// ---------------------------------------------------------------------------
// Kernel 0: convert weights to bf16, transposed to N-major for B-fragments.
//   WqkvT[1536][256] : nn<512 -> Wq[k][nn]; nn in [512,1536) -> Wkv[k][nn-512]
//   WoT  [256][512]  : WoT[nn][k] = Wo[k][nn]
// ---------------------------------------------------------------------------
__global__ void prep_weights(const float* __restrict__ Wq,
                             const float* __restrict__ Wkv,
                             const float* __restrict__ Wo,
                             unsigned short* __restrict__ WqkvT,
                             unsigned short* __restrict__ WoT) {
  int idx = blockIdx.x * blockDim.x + threadIdx.x;
  const int total1 = NQKV * DIM;           // 393216
  const int total2 = DIM * INNER;          // 131072
  if (idx < total1) {
    int nn = idx >> 8;                     // /DIM
    int kk = idx & (DIM - 1);
    float v = (nn < INNER) ? Wq[kk * INNER + nn]
                           : Wkv[kk * (2 * INNER) + (nn - INNER)];
    WqkvT[nn * DIM + kk] = f2bf(v);
  } else if (idx < total1 + total2) {
    int j  = idx - total1;
    int nn = j >> 9;                       // /INNER -> channel 0..255
    int kk = j & (INNER - 1);
    WoT[nn * INNER + kk] = f2bf(Wo[kk * DIM + nn]);
  }
}

// ---------------------------------------------------------------------------
// Kernel 1: fused window attention. grid = 2048 windows, block = 256 threads.
// ---------------------------------------------------------------------------
__global__ __launch_bounds__(256) void local_attn(
    const float* __restrict__ fmap,
    const unsigned short* __restrict__ WqkvT,
    const unsigned short* __restrict__ WoT,
    const float* __restrict__ bo,
    float* __restrict__ out) {

  // ~225 KB LDS total (<= 320 KB/WGP): one workgroup per WGP.
  __shared__ unsigned short sF[MPAD * DIM];            // 32 KB window input (bf16)
  __shared__ unsigned short sQ[HEADS * MPAD * DHEAD];  // 64 KB Q; later attn probs P
  __shared__ unsigned short sK[HEADS * MPAD * DHEAD];  // 64 KB K; later head-concat O
  __shared__ unsigned short sV[HEADS * DHEAD * MPAD];  // 64 KB V^T [head][d][token]
  __shared__ float          sBo[DIM];                  // 1 KB bias cache

  const int tid  = threadIdx.x;
  const int lane = tid & 31;
  const int wave = tid >> 5;
  const int l15  = lane & 15;
  const int hi   = lane >> 4;

  const int wid = blockIdx.x;
  const int bb  = wid >> 6;           // batch
  const int xw  = (wid >> 3) & 7;     // window x
  const int yw  = wid & 7;            // window y

  // ---- Stage 1: load window (49x256 f32) -> bf16 LDS, zero-pad to 64 rows.
  for (int idx = tid; idx < MPAD * DIM; idx += 256) {
    int t = idx >> 8, c = idx & (DIM - 1);
    unsigned short v = 0;
    if (t < PP) {
      int X = xw * PW + t / PW;
      int Y = yw * PW + t % PW;
      v = f2bf(fmap[((bb * 56 + X) * 56 + Y) * DIM + c]);
    }
    sF[idx] = v;
  }
  if (tid < DIM) sBo[tid] = bo[tid];
  __syncthreads();

  // ---- Stage 2: QKV GEMM [64x256] x [256x1536]. mt-outer so each wave's A
  // fragments (8 K-steps, 64 VGPRs) are loaded once and reused for 12 N-tiles.
  for (int mt = 0; mt < 4; ++mt) {
    v16bf afr[8];
    #pragma unroll
    for (int ks = 0; ks < DIM / 32; ++ks)
      afr[ks] = load_a(sF, DIM, mt * 16, ks * 32, lane);

    for (int nt = wave; nt < NQKV / 16; nt += 8) {
      v8f acc = vzero8();
      #pragma unroll
      for (int ks = 0; ks < DIM / 32; ++ks)
        acc = wmma_bf16(afr[ks], load_bt(WqkvT, DIM, nt * 16, ks * 32, lane), acc);

      // Uniform scatter: a 16-wide tile never straddles the Q/K/V (512-wide)
      // or head (64-wide) boundaries, so region/head/channel derive from nt.
      int region = nt >> 5;                 // 0:Q 1:K 2:V
      int hh     = (nt >> 2) & 7;           // head
      int dc     = ((nt & 3) << 4) + l15;   // channel within head
      if (region == 0) {
        #pragma unroll
        for (int r = 0; r < 8; ++r)
          sQ[(hh * MPAD + mt * 16 + r + 8 * hi) * DHEAD + dc] = f2bf(acc[r]);
      } else if (region == 1) {
        #pragma unroll
        for (int r = 0; r < 8; ++r)
          sK[(hh * MPAD + mt * 16 + r + 8 * hi) * DHEAD + dc] = f2bf(acc[r]);
      } else {
        // V transposed [d][token]: rows r, r+1 are contiguous -> packed b32.
        unsigned short* vp = &sV[(hh * DHEAD + dc) * MPAD + mt * 16 + 8 * hi];
        #pragma unroll
        for (int r = 0; r < 8; r += 2) {
          unsigned pk = (unsigned)f2bf(acc[r]) | ((unsigned)f2bf(acc[r + 1]) << 16);
          *(unsigned*)(vp + r) = pk;
        }
      }
    }
  }
  __syncthreads();

  // ---- Stage 3: attention, one wave per head.
  const int h = wave;
  const unsigned short* Qh = sQ + h * MPAD * DHEAD;
  const unsigned short* Kh = sK + h * MPAD * DHEAD;
  const unsigned short* Vh = sV + h * DHEAD * MPAD;
  unsigned short*       Ph = sQ + h * MPAD * MPAD;   // probs alias dead Q[h]

  for (int mt = 0; mt < 4; ++mt) {
    v16bf qa[2];
    #pragma unroll
    for (int ks = 0; ks < DHEAD / 32; ++ks)
      qa[ks] = load_a(Qh, DHEAD, mt * 16, ks * 32, lane);

    v8f d[4];
    #pragma unroll
    for (int nt = 0; nt < 4; ++nt) {
      v8f acc = vzero8();
      #pragma unroll
      for (int ks = 0; ks < DHEAD / 32; ++ks)
        acc = wmma_bf16(qa[ks], load_bt(Kh, DHEAD, nt * 16, ks * 32, lane), acc);
      d[nt] = acc;
    }
    // Row softmax in registers. Element (r): row = mt*16 + r + 8*hi,
    // col = nt*16 + l15. Rows are constant within each 16-lane half, so
    // xor-shuffles with masks 1,2,4,8 reduce across columns only.
    #pragma unroll
    for (int r = 0; r < 8; ++r) {
      float s[4], mx = -1e30f;
      #pragma unroll
      for (int nt = 0; nt < 4; ++nt) {
        int j = nt * 16 + l15;
        float v = d[nt][r] * ATTN_SCALE;
        s[nt] = (j < PP) ? v : -1e30f;       // mask padded keys
        mx = fmaxf(mx, s[nt]);
      }
      #pragma unroll
      for (int m = 1; m < 16; m <<= 1) mx = fmaxf(mx, __shfl_xor(mx, m, 32));
      float e[4], sum = 0.f;
      #pragma unroll
      for (int nt = 0; nt < 4; ++nt) { e[nt] = __expf(s[nt] - mx); sum += e[nt]; }
      #pragma unroll
      for (int m = 1; m < 16; m <<= 1) sum += __shfl_xor(sum, m, 32);
      float inv = 1.0f / sum;
      int row = mt * 16 + r + 8 * hi;
      #pragma unroll
      for (int nt = 0; nt < 4; ++nt)
        Ph[row * MPAD + nt * 16 + l15] = f2bf(e[nt] * inv);
    }
  }

  // ---- Stage 3b: out = P (64x64) x V (64x64), accumulate fully in registers.
  v8f o[16];
  for (int mt = 0; mt < 4; ++mt) {
    v16bf pa[2];
    #pragma unroll
    for (int ks = 0; ks < MPAD / 32; ++ks)
      pa[ks] = load_a(Ph, MPAD, mt * 16, ks * 32, lane);
    #pragma unroll
    for (int dt = 0; dt < 4; ++dt) {
      v8f acc = vzero8();
      #pragma unroll
      for (int ks = 0; ks < MPAD / 32; ++ks)
        acc = wmma_bf16(pa[ks], load_bt(Vh, MPAD, dt * 16, ks * 32, lane), acc);
      o[mt * 4 + dt] = acc;
    }
  }
  __syncthreads();   // everyone done reading sK (K) before it becomes O

  unsigned short* Os = sK;   // head-concat output [64][512] bf16
  for (int mt = 0; mt < 4; ++mt)
    #pragma unroll
    for (int dt = 0; dt < 4; ++dt) {
      v8f acc = o[mt * 4 + dt];
      #pragma unroll
      for (int r = 0; r < 8; ++r) {
        int row = mt * 16 + r + 8 * hi;
        Os[row * INNER + h * DHEAD + dt * 16 + l15] = f2bf(acc[r]);
      }
    }
  __syncthreads();

  // ---- Stage 4: out-proj  [64x512] x [512x256] + bo, scatter to fmap layout.
  for (int t4 = wave; t4 < 4 * (DIM / 16); t4 += 8) {
    int mt = t4 >> 4;
    int nt = t4 & 15;
    v8f acc = vzero8();
    #pragma unroll
    for (int ks = 0; ks < INNER / 32; ++ks) {
      v16bf a = load_a(Os, INNER, mt * 16, ks * 32, lane);
      v16bf b = load_bt(WoT, INNER, nt * 16, ks * 32, lane);
      acc = wmma_bf16(a, b, acc);
    }
    #pragma unroll
    for (int r = 0; r < 8; ++r) {
      int tok = mt * 16 + r + 8 * hi;
      if (tok < PP) {
        int c = nt * 16 + l15;
        int X = xw * PW + tok / PW;
        int Y = yw * PW + tok % PW;
        out[((bb * 56 + X) * 56 + Y) * DIM + c] = acc[r] + sBo[c];
      }
    }
  }
}

// ---------------------------------------------------------------------------
extern "C" void kernel_launch(void* const* d_in, const int* in_sizes, int n_in,
                              void* d_out, int out_size, void* d_ws, size_t ws_size,
                              hipStream_t stream) {
  (void)in_sizes; (void)n_in; (void)out_size; (void)ws_size;
  const float* fmap = (const float*)d_in[0];
  const float* Wq   = (const float*)d_in[1];
  const float* Wkv  = (const float*)d_in[2];
  const float* Wo   = (const float*)d_in[3];
  const float* bo   = (const float*)d_in[4];

  // Workspace: bf16 transposed weights (1 MB total).
  unsigned short* WqkvT = (unsigned short*)d_ws;          // 1536*256*2 = 768 KB
  unsigned short* WoT   = WqkvT + NQKV * DIM;             //  256*512*2 = 256 KB

  const int prepElems = NQKV * DIM + DIM * INNER;         // 524288
  prep_weights<<<(prepElems + 255) / 256, 256, 0, stream>>>(Wq, Wkv, Wo, WqkvT, WoT);

  local_attn<<<32 * 8 * 8, 256, 0, stream>>>(fmap, WqkvT, WoT, bo, (float*)d_out);
}
